// VectorQuantizerEMA_9045201125930
// MI455X (gfx1250) — matrix operand order repaired and checked by reading, not compiled, not run on hardware
//
#include <hip/hip_runtime.h>
#include <hip/hip_bf16.h>

// ---------------------------------------------------------------------------
// VQ-VAE EMA vector quantizer, fused for MI455X (gfx1250, wave32, WMMA + TDM).
// N=131072 rows, D=64, K=1024 codes. Memory-bound (~160MB total traffic);
// distances computed on the fly with v_wmma_f32_16x16x32_f16; the f16 codebook
// is DMA'd into the 320KB WGP LDS by the Tensor Data Mover with hardware row
// padding (72-half pitch -> bank-conflict-free 16-lane fragment loads).
// ---------------------------------------------------------------------------

typedef __attribute__((ext_vector_type(16))) _Float16 v16h;
typedef __attribute__((ext_vector_type(8)))  _Float16 v8h;
typedef __attribute__((ext_vector_type(8)))  float    v8f;
typedef __attribute__((ext_vector_type(4)))  unsigned int uint4v;
typedef __attribute__((ext_vector_type(8)))  int      int8v;
typedef __attribute__((ext_vector_type(4)))  int      int4v;

#define NROWS 131072
#define DDIM  64
#define KCB   1024
#define EPITCH 72           // f16 LDS row pitch for codebook/A tiles (bank-conflict free)

// LDS layout for argmin kernel (dynamic shared):
//  [0)        f16 codebook  1024 x 72 halves  = 147456 B   (TDM-loaded, HW padded)
//  [147456)   f32 norms     1024              =   4096 B
//  [151552)   f16 A tiles   8 waves x 16 x 72 =  18432 B   -> total 169984 B
#define SM_NRM  147456
#define SM_A    151552
#define SM_TOT  169984

// ---------------------------------------------------------------------------
// Kernel Z: zero atomic accumulators (counts, dw in ws; loss slot in d_out)
// ---------------------------------------------------------------------------
__global__ __launch_bounds__(256) void vq_zero_kernel(float* __restrict__ counts,
                                                      float* __restrict__ dw,
                                                      float* __restrict__ loss) {
  int i = blockIdx.x * 256 + threadIdx.x;
  if (i < KCB) counts[i] = 0.0f;
  if (i < KCB * DDIM) dw[i] = 0.0f;
  if (i == 0) *loss = 0.0f;
}

// ---------------------------------------------------------------------------
// Kernel P: embedding fp32 -> f16, plus per-row ||e||^2 (one wave per code)
// ---------------------------------------------------------------------------
__global__ __launch_bounds__(32) void vq_prep_kernel(const float* __restrict__ emb,
                                                     _Float16* __restrict__ e16,
                                                     float* __restrict__ enorm) {
  int c = blockIdx.x, lane = threadIdx.x;
  float a0 = emb[c * 64 + lane];
  float a1 = emb[c * 64 + 32 + lane];
  e16[c * 64 + lane]      = (_Float16)a0;
  e16[c * 64 + 32 + lane] = (_Float16)a1;
  float s = a0 * a0 + a1 * a1;
  #pragma unroll
  for (int sh = 16; sh > 0; sh >>= 1) s += __shfl_xor(s, sh, 32);
  if (lane == 0) enorm[c] = s;
}

// ---------------------------------------------------------------------------
// Kernel A: fused distance + argmin via WMMA, codebook staged by the TDM.
// 256 threads = 8 waves; each wave owns a 16-row tile of flat (=z_e^T rows),
// sweeps all 64 code tiles (16 codes each), K-dim 64 = 2 WMMA ops per tile.
// ---------------------------------------------------------------------------
__global__ __launch_bounds__(256) void vq_argmin_kernel(const float* __restrict__ z_e,
                                                        const _Float16* __restrict__ e16,
                                                        const float* __restrict__ enorm_g,
                                                        int* __restrict__ idx_out) {
  extern __shared__ char smem[];
  _Float16* e_lds = (_Float16*)smem;                // [1024][72], TDM destination
  float*    n_lds = (float*)(smem + SM_NRM);        // [1024]
  _Float16* a_lds = (_Float16*)(smem + SM_A);       // [8][16][72]

  const int tid = threadIdx.x;
  const int wave = tid >> 5, lane = tid & 31;

  // --- Tensor Data Mover: DMA the whole f16 codebook (1024 rows x 64 halves,
  //     row-major) into LDS, with HW padding of 4 DWORDs after every 32 DWORDs
  //     stored => LDS row pitch 72 halves. Wave 0 issues one descriptor.
  if (wave == 0) {
    unsigned int lds_base = (unsigned int)(uintptr_t)e_lds;          // addr[31:0] = LDS byte offset
    unsigned long long ga = (unsigned long long)(uintptr_t)e16;      // 57-bit global byte address
    uint4v g0;
    g0.x = 1u;                                                       // count=1 (valid user D#)
    g0.y = lds_base;                                                 // lds_addr
    g0.z = (unsigned int)ga;                                         // global_addr[31:0]
    g0.w = (unsigned int)((ga >> 32) & 0x01FFFFFFu) | (2u << 30);    // global_addr[56:32], type=2
    int8v g1;
    g1[0] = (int)((1u << 16)        // data_size = 1 -> 2 bytes
                | (1u << 20)        // pad_enable
                | (4u << 22)        // pad_interval = 4 -> every 32 DWORDs (one 64-half row)
                | (3u << 25));      // pad_amount = 3 -> 4 DWORDs (8 halves) of padding
    g1[1] = (int)(64u << 16);       // [47:32] atomic_barrier=0, [63:48] tensor_dim0.lo = 64
    g1[2] = (int)(1024u << 16);     // tensor_dim0.hi = 0, tensor_dim1.lo = 1024
    g1[3] = (int)(64u << 16);       // tensor_dim1.hi = 0, tile_dim0 = 64
    g1[4] = 1024;                   // tile_dim1 = 1024, tile_dim2 = 0
    g1[5] = 64;                     // tensor_dim0_stride = 64 elements
    g1[6] = 0;                      // stride0.hi = 0, tensor_dim1_stride.lo = 0
    g1[7] = 0;
    int4v gz4 = {0, 0, 0, 0};       // 2D tensor: groups 2/3 unused
    int8v gz8 = {0, 0, 0, 0, 0, 0, 0, 0};
    __builtin_amdgcn_tensor_load_to_lds(g0, g1, gz4, gz4, gz8, 0);
  }

  // Norm table: tiny, regular loads (overlaps with the TDM transfer).
  for (int i = tid; i < KCB; i += 256) n_lds[i] = enorm_g[i];

  // Stage this wave's 16x64 A-tile (transpose from BDHW) into LDS as f16.
  const int n0w = blockIdx.x * 128 + wave * 16;          // 16 consecutive n share (b,h)
  const int bb = n0w >> 12, rem = n0w & 4095, hh = rem >> 6, w0 = rem & 63;
  const long zbase = (long)bb * 262144 + hh * 64 + w0;
  const int m = lane & 15, dpar = lane >> 4;
  _Float16* arow_w = a_lds + (wave * 16 + m) * EPITCH;
  #pragma unroll 4
  for (int d2 = 0; d2 < 64; d2 += 2) {
    int d = d2 + dpar;                                   // lanes 0-15 -> d2, 16-31 -> d2+1
    arow_w[d] = (_Float16)z_e[zbase + (long)d * 4096 + m];
  }
  if (wave == 0) __builtin_amdgcn_s_wait_tensorcnt(0);   // codebook resident in LDS
  __syncthreads();

  // Build A fragments per ISA 16-bit A 16x32 layout:
  //  lane<16 : elems 0..7 = A(m, b0+0..7),  8..15 = A(m, b0+16..23)
  //  lane>=16: elems 0..7 = A(m, b0+8..15), 8..15 = A(m, b0+24..31)
  const _Float16* arow = a_lds + (wave * 16 + m) * EPITCH;
  const int alo = (lane < 16) ? 0 : 8;
  union Frag { v16h v; v8h h[2]; };
  Frag A0, A1;
  A0.h[0] = *(const v8h*)(arow + alo);
  A0.h[1] = *(const v8h*)(arow + alo + 16);
  A1.h[0] = *(const v8h*)(arow + 32 + alo);
  A1.h[1] = *(const v8h*)(arow + 32 + alo + 16);

  // Sweep code tiles; track running (min, argmin) per C element.
  float minv[8];
  int   mini[8];
  #pragma unroll
  for (int r = 0; r < 8; ++r) { minv[r] = 3.4e38f; mini[r] = 0; }

  const int col  = lane & 15;
  const int boff = (lane < 16) ? 0 : 8;    // B 32x16: lane<16 holds K=0..7,16..23, else 8..15,24..31
  #pragma unroll 2
  for (int kt = 0; kt < 64; ++kt) {
    const int c = kt * 16 + col;
    const _Float16* brow = e_lds + c * EPITCH + boff;
    Frag B0, B1;
    B0.h[0] = *(const v8h*)(brow);
    B0.h[1] = *(const v8h*)(brow + 16);
    B1.h[0] = *(const v8h*)(brow + 32);
    B1.h[1] = *(const v8h*)(brow + 48);

    v8f acc = {0.f, 0.f, 0.f, 0.f, 0.f, 0.f, 0.f, 0.f};
    acc = __builtin_amdgcn_wmma_f32_16x16x32_f16(false, A0.v, false, B0.v,
                                                 (short)0, acc, false, false);
    acc = __builtin_amdgcn_wmma_f32_16x16x32_f16(false, A1.v, false, B1.v,
                                                 (short)0, acc, false, false);
    const float nrm = n_lds[c];
    #pragma unroll
    for (int r = 0; r < 8; ++r) {
      float dist = nrm - 2.0f * acc[r];   // ||e||^2 - 2 x.e (||x||^2 row-constant)
      if (dist < minv[r]) { minv[r] = dist; mini[r] = c; }
    }
  }

  // Cross-lane argmin over the 16 columns of each C row (ties -> lowest index).
  #pragma unroll
  for (int r = 0; r < 8; ++r) {
    #pragma unroll
    for (int s = 1; s < 16; s <<= 1) {
      float ov = __shfl_xor(minv[r], s, 32);
      int   oi = __shfl_xor(mini[r], s, 32);
      if (ov < minv[r] || (ov == minv[r] && oi < mini[r])) { minv[r] = ov; mini[r] = oi; }
    }
  }
  if (col == 0) {                          // lanes 0 (rows 0..7) and 16 (rows 8..15)
    int half = lane >> 4;
    #pragma unroll
    for (int r = 0; r < 8; ++r) idx_out[n0w + r + 8 * half] = mini[r];
  }
}

// ---------------------------------------------------------------------------
// Kernel S: counts + dw via LDS float atomics (pitch 65 -> bank = (idx+d)%64,
// conflict-free across lanes), then global fp32 atomic flush. 64 blocks.
// ---------------------------------------------------------------------------
__global__ __launch_bounds__(256) void vq_stats_kernel(const float* __restrict__ z_e,
                                                       const int* __restrict__ idx_g,
                                                       float* __restrict__ counts_g,
                                                       float* __restrict__ dw_g) {
  extern __shared__ char smem2[];
  float* dwl = (float*)smem2;                       // [1024][65]
  float* cnt = (float*)(smem2 + KCB * 65 * 4);      // [1024]
  for (int i = threadIdx.x; i < KCB * 65; i += 256) dwl[i] = 0.0f;
  for (int i = threadIdx.x; i < KCB; i += 256) cnt[i] = 0.0f;
  __syncthreads();

  const int base = blockIdx.x * 2048;               // 64 blocks * 2048 rows = N
  for (int it = 0; it < 8; ++it) {
    int n = base + it * 256 + threadIdx.x;
    int b = n >> 12, rem = n & 4095, h = rem >> 6, w = rem & 63;
    long zoff = (long)b * 262144 + h * 64 + w;
    int idx = idx_g[n];
    atomicAdd(&cnt[idx], 1.0f);
    float* drow = dwl + idx * 65;
    __builtin_prefetch(&z_e[zoff], 0, 1);
    #pragma unroll 8
    for (int d = 0; d < 64; ++d)
      atomicAdd(&drow[d], z_e[zoff + (long)d * 4096]);
  }
  __syncthreads();

  for (int i = threadIdx.x; i < KCB * DDIM; i += 256) {
    int c = i >> 6, d = i & 63;
    atomicAdd(&dw_g[i], dwl[c * 65 + d]);
  }
  for (int i = threadIdx.x; i < KCB; i += 256) atomicAdd(&counts_g[i], cnt[i]);
}

// ---------------------------------------------------------------------------
// Kernel E: EMA update + Laplace smoothing + normalized new embedding (1 block)
// ---------------------------------------------------------------------------
__global__ __launch_bounds__(1024) void vq_ema_kernel(const float* __restrict__ cs,
                                                      const float* __restrict__ ea,
                                                      const float* __restrict__ counts,
                                                      const float* __restrict__ dw,
                                                      float* __restrict__ new_emb,
                                                      float* __restrict__ new_cs,
                                                      float* __restrict__ new_ea) {
  __shared__ float wsum[32];
  __shared__ float smoothed[1024];
  __shared__ float ntot_s;
  const int t = threadIdx.x;
  float ncs = cs[t] * 0.99f + 0.01f * counts[t];
  new_cs[t] = ncs;

  float v = ncs;
  #pragma unroll
  for (int s = 16; s > 0; s >>= 1) v += __shfl_xor(v, s, 32);
  if ((t & 31) == 0) wsum[t >> 5] = v;
  __syncthreads();
  if (t < 32) {
    float x = wsum[t];
    #pragma unroll
    for (int s = 16; s > 0; s >>= 1) x += __shfl_xor(x, s, 32);
    if (t == 0) ntot_s = x;
  }
  __syncthreads();
  const float n = ntot_s;
  smoothed[t] = (ncs + 1e-5f) / (n + 1024.0f * 1e-5f) * n;
  __syncthreads();

  for (int i = t; i < KCB * DDIM; i += 1024) {
    float nea = ea[i] * 0.99f + 0.01f * dw[i];
    new_ea[i] = nea;
    new_emb[i] = nea / smoothed[i >> 6];
  }
}

// ---------------------------------------------------------------------------
// Kernel O: z_q_st (== E_old[idx] numerically), float indices, commitment loss
// ---------------------------------------------------------------------------
__global__ __launch_bounds__(256) void vq_out_kernel(const float* __restrict__ z_e,
                                                     const float* __restrict__ emb,
                                                     const int* __restrict__ idx_g,
                                                     float* __restrict__ zq_out,
                                                     float* __restrict__ idx_f_out,
                                                     float* __restrict__ loss_out) {
  __shared__ float red[8];
  const int n = blockIdx.x * 256 + threadIdx.x;
  const int b = n >> 12, rem = n & 4095, h = rem >> 6, w = rem & 63;
  const long zoff = (long)b * 262144 + h * 64 + w;
  const int idx = idx_g[n];
  idx_f_out[n] = (float)idx;
  const float* erow = emb + idx * 64;

  float lsum = 0.0f;
  #pragma unroll 4
  for (int d = 0; d < 64; ++d) {
    float e = erow[d];                                   // gather, L2-resident
    float z = z_e[zoff + (long)d * 4096];
    zq_out[zoff + (long)d * 4096] = e;                   // coalesced over w
    float df = z - e;
    lsum += df * df;
  }
  #pragma unroll
  for (int s = 16; s > 0; s >>= 1) lsum += __shfl_xor(lsum, s, 32);
  const int lane = threadIdx.x & 31, wid = threadIdx.x >> 5;
  if (lane == 0) red[wid] = lsum;
  __syncthreads();
  if (threadIdx.x < 8) {
    float x = red[threadIdx.x];
    #pragma unroll
    for (int s = 4; s > 0; s >>= 1) x += __shfl_xor(x, s, 8);
    if (threadIdx.x == 0)
      atomicAdd(loss_out, x * (0.25f / 8388608.0f));     // BETA * mean over B*D*H*W
  }
}

// ---------------------------------------------------------------------------
extern "C" void kernel_launch(void* const* d_in, const int* in_sizes, int n_in,
                              void* d_out, int out_size, void* d_ws, size_t ws_size,
                              hipStream_t stream) {
  (void)in_sizes; (void)n_in; (void)out_size; (void)ws_size;
  const float* z_e = (const float*)d_in[0];   // [32,64,64,64]
  const float* emb = (const float*)d_in[1];   // [1024,64]
  const float* cs  = (const float*)d_in[2];   // [1024]
  const float* ea  = (const float*)d_in[3];   // [1024,64]

  char* ws = (char*)d_ws;
  _Float16* e16 = (_Float16*)ws;              // 131072 B
  float* enorm  = (float*)(ws + 131072);      //   4096 B
  int*   idx_i  = (int*)(ws + 135168);        // 524288 B
  float* counts = (float*)(ws + 659456);      //   4096 B
  float* dw     = (float*)(ws + 663552);      // 262144 B  (total < 1 MB)

  float* out      = (float*)d_out;
  float* zq_out   = out;                      // 8388608
  float* idx_out  = out + 8388608;            //  131072
  float* loss_out = out + 8519680;            //       1
  float* nemb_out = out + 8519681;            //   65536
  float* ncs_out  = out + 8585217;            //    1024
  float* nea_out  = out + 8586241;            //   65536

  vq_zero_kernel<<<256, 256, 0, stream>>>(counts, dw, loss_out);
  vq_prep_kernel<<<KCB, 32, 0, stream>>>(emb, e16, enorm);
  vq_argmin_kernel<<<NROWS / 128, 256, SM_TOT, stream>>>(z_e, e16, enorm, idx_i);
  vq_stats_kernel<<<64, 256, KCB * 65 * 4 + KCB * 4, stream>>>(z_e, idx_i, counts, dw);
  vq_ema_kernel<<<1, 1024, 0, stream>>>(cs, ea, counts, dw, nemb_out, ncs_out, nea_out);
  vq_out_kernel<<<NROWS / 256, 256, 0, stream>>>(z_e, emb, idx_i, zq_out, idx_out, loss_out);
}
